// Verification_BiLSTM_CRF_46454366273894
// MI455X (gfx1250) — compile-verified, matrix operand order checked by reading, not run on hardware
//
#include <hip/hip_runtime.h>

typedef _Float16 v16h __attribute__((ext_vector_type(16)));
typedef float    v8f  __attribute__((ext_vector_type(8)));

#define CRF_C     8
#define CRF_START 6
#define CRF_STOP  7
#define CRF_L     2048
#define CRF_B     1024
#define NEG_VAL   (-10000.0f)
#define LOG2E_F   1.4426950408889634f
#define LN2_F     0.6931471805599453f

// Fast hardware transcendentals: v_exp_f32 / v_log_f32 are base-2.
__device__ __forceinline__ float fast_exp2(float x) { return __builtin_amdgcn_exp2f(x); }
__device__ __forceinline__ float fast_log2(float x) { return __builtin_amdgcn_logf(x); }

// Forward scan, 16 batches per wave32. Inner 8x8 log-semiring matvec factored
// into a standard matmul (A = exp(trans), B = exp(scores - max)) executed by
// v_wmma_f32_16x16x32_f16. Gold-path score fused in. Writes (fwd - gold) per batch.
__global__ __launch_bounds__(128) void crf_forward_wmma(
    const float* __restrict__ feats, const float* __restrict__ trans,
    const int*   __restrict__ tags,  const int*   __restrict__ lengths,
    float* __restrict__ partial)
{
    __shared__ float sT[64];
    const int tid = threadIdx.x;
    for (int i = tid; i < 64; i += blockDim.x) sT[i] = trans[i];
    __syncthreads();

    const int  lane = tid & 31;
    const int  wave = (int)((blockIdx.x * blockDim.x + tid) >> 5);
    const int  r    = lane & 15;              // batch slot within wave (mirrored in upper half)
    const int  b    = wave * 16 + r;          // batch index (lanes 16..31 mirror 0..15)
    const bool low  = lane < 16;

    // ---- A matrix: T'[j,k] = exp(trans[j,k]), f16 16x32 layout.
    // Only lanes 0..15, elements 0..7 (K=0..7, M=r<8) are nonzero; all other
    // K/M positions are zero under either 16-bit operand layout interpretation.
    v16h A = {};
    if (low && r < CRF_C) {
        A[0] = (_Float16)fast_exp2(sT[r * 8 + 0] * LOG2E_F);
        A[1] = (_Float16)fast_exp2(sT[r * 8 + 1] * LOG2E_F);
        A[2] = (_Float16)fast_exp2(sT[r * 8 + 2] * LOG2E_F);
        A[3] = (_Float16)fast_exp2(sT[r * 8 + 3] * LOG2E_F);
        A[4] = (_Float16)fast_exp2(sT[r * 8 + 4] * LOG2E_F);
        A[5] = (_Float16)fast_exp2(sT[r * 8 + 5] * LOG2E_F);
        A[6] = (_Float16)fast_exp2(sT[r * 8 + 6] * LOG2E_F);
        A[7] = (_Float16)fast_exp2(sT[r * 8 + 7] * LOG2E_F);
    }

    // ---- scan state: scores for this lane's batch
    float s0 = NEG_VAL, s1 = NEG_VAL, s2 = NEG_VAL, s3 = NEG_VAL;
    float s4 = NEG_VAL, s5 = NEG_VAL, s6 = 0.0f /*START*/, s7 = NEG_VAL;

    const int len = lengths[b];
    int lmax = len;
#pragma unroll
    for (int off = 16; off > 0; off >>= 1) {
        int o = __shfl_xor(lmax, off, 32);
        lmax = lmax > o ? lmax : o;
    }
    // lmax is identical in every lane; make it a wave-uniform SGPR so the scan
    // loop uses scalar control flow and EXEC is provably all-1s at the WMMA.
    lmax = __builtin_amdgcn_readfirstlane(lmax);

    float gold = 0.0f;
    int   prev = CRF_START;

    for (int t = 0; t < lmax; ++t) {
        const size_t  base = (size_t)t * CRF_B + b;
        const float4* fp   = (const float4*)feats + base * 2;
        const float4  fa   = fp[0];               // emit[0..3]
        const float4  fb   = fp[1];               // emit[4..7]
        const int     tag  = tags[base];
        const bool    msk  = t < len;

        if (t + 4 < lmax) {
            __builtin_prefetch((const void*)((const float4*)feats +
                               (base + 4 * (size_t)CRF_B) * 2), 0, 1);
        }

        // ---- gold path (fused): emit[tag] selected from registers, trans from LDS
        {
            float e01 = (tag & 1) ? fa.y : fa.x;
            float e23 = (tag & 1) ? fa.w : fa.z;
            float e45 = (tag & 1) ? fb.y : fb.x;
            float e67 = (tag & 1) ? fb.w : fb.z;
            float eL  = (tag & 2) ? e23 : e01;
            float eH  = (tag & 2) ? e67 : e45;
            float emit_sc = (tag & 4) ? eH : eL;
            float step_sc = emit_sc + sT[tag * 8 + prev];
            gold += msk ? step_sc : 0.0f;
            prev  = msk ? tag : prev;
        }

        // ---- forward step: v = exp(s - m); D = T' * v (WMMA); s' = emit + m + log(D)
        float m = fmaxf(fmaxf(fmaxf(s0, s1), fmaxf(s2, s3)),
                        fmaxf(fmaxf(s4, s5), fmaxf(s6, s7)));
        float e0 = fast_exp2((s0 - m) * LOG2E_F);
        float e1 = fast_exp2((s1 - m) * LOG2E_F);
        float e2 = fast_exp2((s2 - m) * LOG2E_F);
        float e3 = fast_exp2((s3 - m) * LOG2E_F);
        float e4 = fast_exp2((s4 - m) * LOG2E_F);
        float e5 = fast_exp2((s5 - m) * LOG2E_F);
        float e6 = fast_exp2((s6 - m) * LOG2E_F);
        float e7 = fast_exp2((s7 - m) * LOG2E_F);

        v16h Bv = {};
        Bv[0] = (_Float16)(low ? e0 : 0.0f);
        Bv[1] = (_Float16)(low ? e1 : 0.0f);
        Bv[2] = (_Float16)(low ? e2 : 0.0f);
        Bv[3] = (_Float16)(low ? e3 : 0.0f);
        Bv[4] = (_Float16)(low ? e4 : 0.0f);
        Bv[5] = (_Float16)(low ? e5 : 0.0f);
        Bv[6] = (_Float16)(low ? e6 : 0.0f);
        Bv[7] = (_Float16)(low ? e7 : 0.0f);

        v8f cz = {};
        v8f d = __builtin_amdgcn_wmma_f32_16x16x32_f16(
                    false, A, false, Bv, (short)0, cz, false, false);

        float n0 = fa.x + m + LN2_F * fast_log2(d[0]);
        float n1 = fa.y + m + LN2_F * fast_log2(d[1]);
        float n2 = fa.z + m + LN2_F * fast_log2(d[2]);
        float n3 = fa.w + m + LN2_F * fast_log2(d[3]);
        float n4 = fb.x + m + LN2_F * fast_log2(d[4]);
        float n5 = fb.y + m + LN2_F * fast_log2(d[5]);
        float n6 = fb.z + m + LN2_F * fast_log2(d[6]);
        float n7 = fb.w + m + LN2_F * fast_log2(d[7]);

        s0 = msk ? n0 : s0;  s1 = msk ? n1 : s1;
        s2 = msk ? n2 : s2;  s3 = msk ? n3 : s3;
        s4 = msk ? n4 : s4;  s5 = msk ? n5 : s5;
        s6 = msk ? n6 : s6;  s7 = msk ? n7 : s7;
    }

    // ---- forward score: logsumexp(scores + trans[STOP, :])
    float t0 = s0 + sT[CRF_STOP * 8 + 0];
    float t1 = s1 + sT[CRF_STOP * 8 + 1];
    float t2 = s2 + sT[CRF_STOP * 8 + 2];
    float t3 = s3 + sT[CRF_STOP * 8 + 3];
    float t4 = s4 + sT[CRF_STOP * 8 + 4];
    float t5 = s5 + sT[CRF_STOP * 8 + 5];
    float t6 = s6 + sT[CRF_STOP * 8 + 6];
    float t7 = s7 + sT[CRF_STOP * 8 + 7];
    float m2 = fmaxf(fmaxf(fmaxf(t0, t1), fmaxf(t2, t3)),
                     fmaxf(fmaxf(t4, t5), fmaxf(t6, t7)));
    float sum = fast_exp2((t0 - m2) * LOG2E_F) + fast_exp2((t1 - m2) * LOG2E_F)
              + fast_exp2((t2 - m2) * LOG2E_F) + fast_exp2((t3 - m2) * LOG2E_F)
              + fast_exp2((t4 - m2) * LOG2E_F) + fast_exp2((t5 - m2) * LOG2E_F)
              + fast_exp2((t6 - m2) * LOG2E_F) + fast_exp2((t7 - m2) * LOG2E_F);
    float fwd  = m2 + LN2_F * fast_log2(sum);
    float gtot = gold + sT[CRF_STOP * 8 + prev];   // prev == tags[len-1]

    if (low) partial[b] = fwd - gtot;
}

__global__ __launch_bounds__(256) void crf_mean(const float* __restrict__ partial,
                                                float* __restrict__ out)
{
    __shared__ float buf[256];
    const int tid = threadIdx.x;
    float s = 0.0f;
    for (int i = tid; i < CRF_B; i += 256) s += partial[i];
    buf[tid] = s;
    __syncthreads();
    for (int k = 128; k > 0; k >>= 1) {
        if (tid < k) buf[tid] += buf[tid + k];
        __syncthreads();
    }
    if (tid == 0) out[0] = buf[0] * (1.0f / (float)CRF_B);
}

extern "C" void kernel_launch(void* const* d_in, const int* in_sizes, int n_in,
                              void* d_out, int out_size, void* d_ws, size_t ws_size,
                              hipStream_t stream)
{
    (void)in_sizes; (void)n_in; (void)out_size; (void)ws_size;
    const float* feats   = (const float*)d_in[0];
    const float* trans   = (const float*)d_in[1];
    const int*   tags    = (const int*)d_in[2];
    const int*   lengths = (const int*)d_in[3];
    float*       partial = (float*)d_ws;           // B floats of scratch
    float*       out     = (float*)d_out;

    // 64 wave32s total: 16 blocks x 128 threads (4 waves/block), 16 batches/wave.
    crf_forward_wmma<<<dim3(16), dim3(128), 0, stream>>>(feats, trans, tags, lengths, partial);
    crf_mean<<<dim3(1), dim3(256), 0, stream>>>(partial, out);
}
